// ZAGRNN_46961172415106
// MI455X (gfx1250) — matrix operand order repaired and checked by reading, not compiled, not run on hardware
//
#include <hip/hip_runtime.h>
#include <math.h>

typedef __bf16 bf16;
typedef long long ll;
typedef __attribute__((ext_vector_type(16))) __bf16 v16bf;
typedef __attribute__((ext_vector_type(8)))  __bf16 v8bf;
typedef __attribute__((ext_vector_type(8)))  float  v8f;

constexpr int cB = 8, cL = 2000, cE = 300, cNL = 8922, cLW = 10;
constexpr int cF = 200, cK = 10, cH = 200, cFEAT = 500, cLP = 1991;  // L-K+1
// padded leading dims (multiples of 32 elements = 64B, pads are zeroed)
constexpr int pE = 320, pF = 224, pNL = 8928, pLP = 2016;

// ---------------------------------------------------------------------------
// gfx1250 async global->LDS staging (16B per lane), guarded; fallback = v8bf
// builtin signature: (v4i as1*, v4i as3*, imm offset, imm cpol)
// ---------------------------------------------------------------------------
#if defined(__HIP_DEVICE_COMPILE__) && defined(__gfx1250__) && \
    __has_builtin(__builtin_amdgcn_global_load_async_to_lds_b128)
#define USE_ASYNC_LDS 1
typedef int v4i __attribute__((vector_size(16)));
typedef __attribute__((address_space(1))) v4i as1_v4i;
typedef __attribute__((address_space(3))) v4i as3_v4i;
#else
#define USE_ASYNC_LDS 0
#endif

__device__ __forceinline__ void stage16(const bf16* g, bf16* l) {
#if USE_ASYNC_LDS
  __builtin_amdgcn_global_load_async_to_lds_b128((as1_v4i*)g, (as3_v4i*)l, 0, 0);
#else
  *(v8bf*)l = *(const v8bf*)g;
#endif
}

template <int N>
__device__ __forceinline__ void wait_async() {
#if USE_ASYNC_LDS
#if __has_builtin(__builtin_amdgcn_s_wait_asynccnt)
  __builtin_amdgcn_s_wait_asynccnt(N);
#else
  asm volatile("s_wait_asynccnt %0" ::"i"(N) : "memory");
#endif
#endif
}

// ---------------------------------------------------------------------------
// WMMA fragment helpers (CDNA5 wave32 layouts, 16x16x32 bf16)
// ---------------------------------------------------------------------------
__device__ __forceinline__ v16bf lds_a_frag(const bf16* base, int lane, int stride) {
  int m = lane & 15, h = lane >> 4;
  const bf16* row = base + m * stride;
  v8bf lo = *(const v8bf*)(row + 8 * h);
  v8bf hi = *(const v8bf*)(row + 16 + 8 * h);
  return __builtin_shufflevector(lo, hi, 0,1,2,3,4,5,6,7,8,9,10,11,12,13,14,15);
}
__device__ __forceinline__ v16bf lds_b_frag(const bf16* base, int lane, int stride) {
  int n = lane & 15, h = lane >> 4;
  const bf16* row = base + n * stride;
  v8bf lo = *(const v8bf*)(row + 8 * h);
  v8bf hi = *(const v8bf*)(row + 16 + 8 * h);
  return __builtin_shufflevector(lo, hi, 0,1,2,3,4,5,6,7,8,9,10,11,12,13,14,15);
}
__device__ __forceinline__ v8f wmma_bf16(v16bf a, v16bf b, v8f c) {
  return __builtin_amdgcn_wmma_f32_16x16x32_bf16(false, a, false, b, (short)0, c,
                                                 false, false);
}

// ---------------------------------------------------------------------------
// prep kernels
// ---------------------------------------------------------------------------
__global__ void cvt_adj_kernel(const float* adj, bf16* out) {
  ll n = (ll)cNL * pNL;
  for (ll i = (ll)blockIdx.x * blockDim.x + threadIdx.x; i < n;
       i += (ll)gridDim.x * blockDim.x) {
    ll r = i / pNL;
    int c = (int)(i - r * pNL);
    out[i] = (c < cNL) ? (bf16)adj[r * cNL + c] : (bf16)0.0f;
  }
}

// out[r*ldo + c] = (c<C) ? in[c*ldi + r] : 0, for r<R, c<ldo
__global__ void transpose_pad_bf16_kernel(const float* in, bf16* out, int R,
                                          int C, int ldi, int ldo) {
  ll n = (ll)R * ldo;
  for (ll i = (ll)blockIdx.x * blockDim.x + threadIdx.x; i < n;
       i += (ll)gridDim.x * blockDim.x) {
    int r = (int)(i / ldo), c = (int)(i - (ll)r * ldo);
    out[i] = (c < C) ? (bf16)in[(ll)c * ldi + r] : (bf16)0.0f;
  }
}

// Wc [F,E,K] -> WcT [K][F][pE]  (per-tap planes, pad cols zero)
__global__ void wcperm_kernel(const float* Wc, bf16* WcT) {
  ll n = (ll)cK * cF * pE;
  for (ll i = (ll)blockIdx.x * blockDim.x + threadIdx.x; i < n;
       i += (ll)gridDim.x * blockDim.x) {
    int k = (int)(i / (cF * pE));
    int rem = (int)(i - (ll)k * (cF * pE));
    int f = rem / pE, e = rem - f * pE;
    WcT[i] = (e < cE) ? (bf16)Wc[((ll)f * cE + e) * cK + k] : (bf16)0.0f;
  }
}

// xe[b,l,:] = word_emb[x[b,l]] * mask[b,l] -> bf16 [B*L, pE] (pad zero)
__global__ void xe_build_kernel(const int* x, const float* mask,
                                const float* wemb, bf16* xe) {
  ll n = (ll)cB * cL * pE;
  for (ll i = (ll)blockIdx.x * blockDim.x + threadIdx.x; i < n;
       i += (ll)gridDim.x * blockDim.x) {
    ll bl = i / pE;
    int e = (int)(i - bl * pE);
    bf16 v = (bf16)0.0f;
    if (e < cE) {
      int tok = x[bl];
      v = (bf16)(wemb[(ll)tok * cE + e] * mask[bl]);
    }
    xe[i] = v;
  }
}

// avg label embedding -> graph_emb[:, :E] (f32), avg [NL,pE], avgT [E,pNL]
__global__ void label_avg_kernel(const int* lidx, const float* lmask,
                                 const float* wemb, float* gemb, bf16* avg_b,
                                 bf16* avgT_b) {
  ll n = (ll)cNL * cE;
  for (ll i = (ll)blockIdx.x * blockDim.x + threadIdx.x; i < n;
       i += (ll)gridDim.x * blockDim.x) {
    int nl = (int)(i / cE);
    int e = (int)(i - (ll)nl * cE);
    float s = 0.f, d = 0.f;
#pragma unroll
    for (int w = 0; w < cLW; ++w) {
      float mw = lmask[nl * cLW + w];
      s += wemb[(ll)lidx[nl * cLW + w] * cE + e] * mw;
      d += mw;
    }
    float a = s / d;
    gemb[(ll)nl * cFEAT + e] = a;
    avg_b[(ll)nl * pE + e] = (bf16)a;
    avgT_b[(ll)e * pNL + nl] = (bf16)a;
  }
}

// zero pad columns [from, stride) of a [rows x stride] bf16 buffer
__global__ void zpad_kernel(bf16* buf, ll rows, int stride, int from) {
  int w = stride - from;
  ll n = rows * w;
  for (ll i = (ll)blockIdx.x * blockDim.x + threadIdx.x; i < n;
       i += (ll)gridDim.x * blockDim.x) {
    ll r = i / w;
    int c = from + (int)(i - r * w);
    buf[r * stride + c] = (bf16)0.0f;
  }
}

// ---------------------------------------------------------------------------
// generic NT GEMM: C[m,n] = sum_k A[m,k]*Bt[n,k]; Kd multiple of 32, pads zero.
// Branchless staging (row indices clamped; OOB rows/cols masked at store).
// Double-buffered async global->LDS pipeline.
// epi 0: acc / rowscale[m] -> outb ; epi 1: relu(acc+bias[n]) -> outf ;
// epi 2: tanh(acc+bias[n]) -> outb.  block=128 (4 waves), tile 64x64x32.
// ---------------------------------------------------------------------------
__global__ void __launch_bounds__(128)
gemm_nt_kernel(const bf16* A, ll lda, const bf16* Bt, ll ldb, int M, int N,
               int Kd, int epi, const float* rowscale, const float* bias,
               float* outf, ll ldo_f, bf16* outb, ll ldo_b) {
  __shared__ __align__(16) bf16 At[2][64 * 40];
  __shared__ __align__(16) bf16 Bs[2][64 * 40];
  int tid = threadIdx.x, lane = tid & 31, wave = tid >> 5;
  ll m0 = (ll)blockIdx.x * 64;
  ll n0 = (ll)blockIdx.y * 64;
  v8f z8 = {0.f, 0.f, 0.f, 0.f, 0.f, 0.f, 0.f, 0.f};
  v8f acc[4] = {z8, z8, z8, z8};
  int nk = Kd >> 5;

  auto stage = [&](int kc, int buf) {
    ll k0 = (ll)kc << 5;
#pragma unroll
    for (int i = tid; i < 512; i += 128) {  // 256 A + 256 B 16B-chunks
      int t = i & 255;
      int r = t >> 2, j = (t & 3) << 3;
      if (i < 256) {
        ll mm = m0 + r;
        if (mm >= M) mm = M - 1;
        stage16(A + mm * lda + k0 + j, At[buf] + r * 40 + j);
      } else {
        ll nn = n0 + r;
        if (nn >= N) nn = N - 1;
        stage16(Bt + nn * ldb + k0 + j, Bs[buf] + r * 40 + j);
      }
    }
  };

  stage(0, 0);
  for (int kc = 0; kc < nk; ++kc) {
    int cur = kc & 1;
    if (kc + 1 < nk) {
      stage(kc + 1, cur ^ 1);
      wait_async<4>();  // in-order: chunk kc's 4 copies have landed
    } else {
      wait_async<0>();
    }
    __syncthreads();
    v16bf a = lds_a_frag(At[cur] + wave * 16 * 40, lane, 40);
#pragma unroll
    for (int j = 0; j < 4; ++j) {
      v16bf b = lds_b_frag(Bs[cur] + j * 16 * 40, lane, 40);
      acc[j] = wmma_bf16(a, b, acc[j]);
    }
    __syncthreads();  // all reads done before buffer is restaged
  }
  int col = lane & 15, half = lane >> 4;
#pragma unroll
  for (int j = 0; j < 4; ++j) {
    ll nn = n0 + j * 16 + col;
#pragma unroll
    for (int r = 0; r < 8; ++r) {
      ll mm = m0 + wave * 16 + r + 8 * half;
      if (mm >= M || nn >= N) continue;
      float v = acc[j][r];
      if (epi == 0) {
        outb[mm * ldo_b + nn] = (bf16)(v / rowscale[mm]);
      } else if (epi == 1) {
        outf[mm * ldo_f + nn] = fmaxf(v + bias[nn], 0.0f);
      } else {
        outb[mm * ldo_b + nn] = (bf16)tanhf(v + bias[nn]);
      }
    }
  }
}

// ---------------------------------------------------------------------------
// conv: c[b,lp,f] = relu(sum_k sum_e xe[b,lp+k,e]*Wc[f,e,k] + bc[f])
// 10 accumulated per-tap GEMMs, flattened into one double-buffered pipeline.
// Writes c [.,pF] and cT [.,pLP] (pads written zero).
// ---------------------------------------------------------------------------
__global__ void __launch_bounds__(128)
conv_gemm_kernel(const bf16* xe, const bf16* WcT, const float* bc, bf16* c_out,
                 bf16* cT_out) {
  __shared__ __align__(16) bf16 At[2][64 * 40];
  __shared__ __align__(16) bf16 Bs[2][64 * 40];
  int tid = threadIdx.x, lane = tid & 31, wave = tid >> 5;
  int b = blockIdx.z;
  int lp0 = blockIdx.x * 64, f0 = blockIdx.y * 64;
  v8f z8 = {0.f, 0.f, 0.f, 0.f, 0.f, 0.f, 0.f, 0.f};
  v8f acc[4] = {z8, z8, z8, z8};
  const int NEC = pE / 32;        // 10 e-chunks per tap
  const int NCH = cK * NEC;       // 100 chunks total
  const bf16* xbase = xe + (ll)b * cL * pE;

  auto stage = [&](int tk, int buf) {
    int k = tk / NEC, e0 = (tk - (tk / NEC) * NEC) << 5;
    const bf16* wbase = WcT + (ll)k * cF * pE;
#pragma unroll
    for (int i = tid; i < 512; i += 128) {
      int t = i & 255;
      int r = t >> 2, j = (t & 3) << 3;
      if (i < 256) {
        int row = lp0 + r + k;
        if (row > cL - 1) row = cL - 1;  // clamp; garbage rows masked at store
        stage16(xbase + (ll)row * pE + e0 + j, At[buf] + r * 40 + j);
      } else {
        int f = f0 + r;
        if (f > cF - 1) f = cF - 1;
        stage16(wbase + (ll)f * pE + e0 + j, Bs[buf] + r * 40 + j);
      }
    }
  };

  stage(0, 0);
  for (int tk = 0; tk < NCH; ++tk) {
    int cur = tk & 1;
    if (tk + 1 < NCH) {
      stage(tk + 1, cur ^ 1);
      wait_async<4>();
    } else {
      wait_async<0>();
    }
    __syncthreads();
    v16bf a = lds_a_frag(At[cur] + wave * 16 * 40, lane, 40);
#pragma unroll
    for (int j = 0; j < 4; ++j) {
      v16bf bb = lds_b_frag(Bs[cur] + j * 16 * 40, lane, 40);
      acc[j] = wmma_bf16(a, bb, acc[j]);
    }
    __syncthreads();
  }
  int col = lane & 15, half = lane >> 4;
#pragma unroll
  for (int j = 0; j < 4; ++j) {
    int f = f0 + j * 16 + col;
#pragma unroll
    for (int r = 0; r < 8; ++r) {
      int lp = lp0 + wave * 16 + r + 8 * half;
      float v = 0.0f;
      if (lp < cLP && f < cF) v = fmaxf(acc[j][r] + bc[f], 0.0f);
      if (lp < cLP && f < pF)
        c_out[((ll)b * cLP + lp) * pF + f] = (bf16)v;
      if (f < cF && lp < pLP)
        cT_out[((ll)b * cF + f) * pLP + lp] = (bf16)v;
    }
  }
}

// ---------------------------------------------------------------------------
// flash-style label attention: per (b, 32-label tile), stream over Lp.
// Q = avg [NL,pE], Km = pj [B*Lp,pE], Vt = cT [B*F,pLP]; out attn [B*NL,pF]
// 2 waves/WG; LDS = 60,160 B. Clamped-row staging; cols >= Lp masked in
// softmax (P=0), garbage rows/cols masked at store.
// ---------------------------------------------------------------------------
__global__ void __launch_bounds__(64)
attn_flash_kernel(const bf16* Q, const bf16* Km, const bf16* Vt, bf16* Oc) {
  __shared__ __align__(16) bf16 Qt[32 * pE];
  __shared__ __align__(16) bf16 Kt[32 * pE];
  __shared__ __align__(16) bf16 Vs[208 * 40];
  __shared__ __align__(16) bf16 Pt[2 * 16 * 40];
  int tid = threadIdx.x, lane = tid & 31, wave = tid >> 5;
  int b = blockIdx.y;
  int n0 = blockIdx.x * 32;
  // stage Q tile once: 32 rows x 40 16B-chunks
  for (int i = tid; i < 32 * 40; i += 64) {
    int r = i / 40, j = (i - (i / 40) * 40) << 3;
    int nlab = n0 + r;
    if (nlab > cNL - 1) nlab = cNL - 1;
    stage16(Q + (ll)nlab * pE + j, Qt + r * pE + j);
  }
  wait_async<0>();
  __syncthreads();

  v8f z8 = {0.f, 0.f, 0.f, 0.f, 0.f, 0.f, 0.f, 0.f};
  v8f O[13];
#pragma unroll
  for (int ft = 0; ft < 13; ++ft) O[ft] = z8;
  float mrun[8], lrun[8];
#pragma unroll
  for (int r = 0; r < 8; ++r) { mrun[r] = -1e30f; lrun[r] = 0.0f; }
  int col = lane & 15, half = lane >> 4;

  int nch = (cLP + 31) / 32;  // 63
  for (int ch = 0; ch < nch; ++ch) {
    int l0 = ch * 32;
    __syncthreads();  // protect K/V restage vs previous reads
    for (int i = tid; i < 32 * 40; i += 64) {  // K rows (l), 320 e each
      int r = i / 40, j = (i - (i / 40) * 40) << 3;
      int l = l0 + r;
      if (l > cLP - 1) l = cLP - 1;
      stage16(Km + ((ll)b * cLP + l) * pE + j, Kt + r * pE + j);
    }
    for (int i = tid; i < 208 * 4; i += 64) {  // V rows (f), 32 l each
      int f = i >> 2, j = (i & 3) << 3;
      int fc = f;
      if (fc > cF - 1) fc = cF - 1;
      stage16(Vt + ((ll)b * cF + fc) * pLP + l0 + j, Vs + f * 40 + j);
    }
    wait_async<0>();
    __syncthreads();

    v8f s0 = z8, s1 = z8;
#pragma unroll
    for (int ec = 0; ec < pE / 32; ++ec) {
      v16bf a = lds_a_frag(Qt + wave * 16 * pE + ec * 32, lane, pE);
      v16bf b0 = lds_b_frag(Kt + ec * 32, lane, pE);
      v16bf b1 = lds_b_frag(Kt + 16 * pE + ec * 32, lane, pE);
      s0 = wmma_bf16(a, b0, s0);
      s1 = wmma_bf16(a, b1, s1);
    }
    bool v0ok = (l0 + col) < cLP;
    bool v1ok = (l0 + 16 + col) < cLP;
#pragma unroll
    for (int r = 0; r < 8; ++r) {
      float a0 = v0ok ? s0[r] : -3.0e38f;
      float a1 = v1ok ? s1[r] : -3.0e38f;
      float mx = fmaxf(a0, a1);
      mx = fmaxf(mx, __shfl_xor(mx, 1));
      mx = fmaxf(mx, __shfl_xor(mx, 2));
      mx = fmaxf(mx, __shfl_xor(mx, 4));
      mx = fmaxf(mx, __shfl_xor(mx, 8));
      float mnew = fmaxf(mrun[r], mx);
      float scale = __expf(mrun[r] - mnew);
      mrun[r] = mnew;
      lrun[r] *= scale;
#pragma unroll
      for (int ft = 0; ft < 13; ++ft) O[ft][r] *= scale;
      float p0 = v0ok ? __expf(s0[r] - mnew) : 0.0f;
      float p1 = v1ok ? __expf(s1[r] - mnew) : 0.0f;
      s0[r] = p0;
      s1[r] = p1;
      float rs = p0 + p1;
      rs += __shfl_xor(rs, 1);
      rs += __shfl_xor(rs, 2);
      rs += __shfl_xor(rs, 4);
      rs += __shfl_xor(rs, 8);
      lrun[r] += rs;
    }
    // bounce P through LDS into A-fragment layout
    bf16* pw = Pt + wave * 16 * 40;
#pragma unroll
    for (int r = 0; r < 8; ++r) {
      int row = r + 8 * half;
      pw[row * 40 + col] = (bf16)s0[r];
      pw[row * 40 + 16 + col] = (bf16)s1[r];
    }
    __syncthreads();
    v16bf ap = lds_a_frag(pw, lane, 40);
#pragma unroll
    for (int ft = 0; ft < 13; ++ft) {
      v16bf bv = lds_b_frag(Vs + ft * 16 * 40, lane, 40);
      O[ft] = wmma_bf16(ap, bv, O[ft]);
    }
  }
#pragma unroll
  for (int r = 0; r < 8; ++r) {
    float inv = 1.0f / lrun[r];
    int nlab = n0 + wave * 16 + r + 8 * half;
    if (nlab < cNL) {
#pragma unroll
      for (int ft = 0; ft < 13; ++ft) {
        int f = ft * 16 + col;
        if (f < cF)
          Oc[((ll)b * cNL + nlab) * pF + f] = (bf16)(O[ft][r] * inv);
      }
    }
  }
}

// ---------------------------------------------------------------------------
// proj = relu(attn@Wo + bo); logits[b,n] = sum_j proj * graph_emb[n,j]
// Full N looped inside the block -> exact row sums, no atomics.
// ---------------------------------------------------------------------------
__global__ void __launch_bounds__(128)
proj_logits_kernel(const bf16* A, const bf16* Bt, const float* bo,
                   const float* ge, float* logits) {
  __shared__ __align__(16) bf16 At[64 * 40];
  __shared__ __align__(16) bf16 Bs[64 * 40];
  const int M = cB * cNL, N = cFEAT, Kd = pF;
  int tid = threadIdx.x, lane = tid & 31, wave = tid >> 5;
  ll m0 = (ll)blockIdx.x * 64;
  int col = lane & 15, half = lane >> 4;
  v8f z8 = {0.f, 0.f, 0.f, 0.f, 0.f, 0.f, 0.f, 0.f};
  float rsum[8] = {0.f, 0.f, 0.f, 0.f, 0.f, 0.f, 0.f, 0.f};
  for (int nt = 0; nt < 8; ++nt) {
    int n0 = nt * 64;
    v8f acc[4] = {z8, z8, z8, z8};
    for (int kc = 0; kc < Kd / 32; ++kc) {
      int k0 = kc << 5;
      __syncthreads();
#pragma unroll
      for (int i = tid; i < 512; i += 128) {
        int t = i & 255;
        int r = t >> 2, j = (t & 3) << 3;
        if (i < 256) {
          ll mm = m0 + r;
          if (mm >= M) mm = M - 1;
          stage16(A + mm * pF + k0 + j, At + r * 40 + j);
        } else {
          int nn = n0 + r;
          if (nn >= N) nn = N - 1;
          stage16(Bt + (ll)nn * pF + k0 + j, Bs + r * 40 + j);
        }
      }
      wait_async<0>();
      __syncthreads();
      v16bf a = lds_a_frag(At + wave * 16 * 40, lane, 40);
#pragma unroll
      for (int j = 0; j < 4; ++j) {
        v16bf b = lds_b_frag(Bs + j * 16 * 40, lane, 40);
        acc[j] = wmma_bf16(a, b, acc[j]);
      }
    }
#pragma unroll
    for (int j = 0; j < 4; ++j) {
      int nn = n0 + j * 16 + col;
#pragma unroll
      for (int r = 0; r < 8; ++r) {
        ll mm = m0 + wave * 16 + r + 8 * half;
        if (mm < M && nn < N) {
          int lbl = (int)(mm % cNL);
          float v = fmaxf(acc[j][r] + bo[nn], 0.0f) * ge[(ll)lbl * cFEAT + nn];
          rsum[r] += v;
        }
      }
    }
  }
#pragma unroll
  for (int r = 0; r < 8; ++r) {
    float s = rsum[r];
    s += __shfl_xor(s, 1);
    s += __shfl_xor(s, 2);
    s += __shfl_xor(s, 4);
    s += __shfl_xor(s, 8);
    if (col == 0) {
      ll mm = m0 + wave * 16 + r + 8 * half;
      if (mm < M) logits[mm] = s;
    }
  }
}

// single-block deterministic BCE reduction
__global__ void __launch_bounds__(256)
loss_kernel(const float* logits, const float* y, float* loss) {
  __shared__ float red[256];
  ll n = (ll)cB * cNL;
  float s = 0.f;
  for (ll i = threadIdx.x; i < n; i += 256) {
    float l = logits[i];
    s += fmaxf(l, 0.0f) - l * y[i] + log1pf(__expf(-fabsf(l)));
  }
  red[threadIdx.x] = s;
  __syncthreads();
  for (int o = 128; o > 0; o >>= 1) {
    if (threadIdx.x < o) red[threadIdx.x] += red[threadIdx.x + o];
    __syncthreads();
  }
  if (threadIdx.x == 0) *loss = red[0] * (1.0f / cB);
}

// ---------------------------------------------------------------------------
extern "C" void kernel_launch(void* const* d_in, const int* in_sizes, int n_in,
                              void* d_out, int out_size, void* d_ws,
                              size_t ws_size, hipStream_t stream) {
  (void)in_sizes; (void)n_in; (void)out_size; (void)ws_size;
  const int*   x     = (const int*)d_in[0];
  const float* y     = (const float*)d_in[1];
  const float* mask  = (const float*)d_in[2];
  const float* wemb  = (const float*)d_in[3];
  const int*   lidx  = (const int*)d_in[4];
  const float* lmask = (const float*)d_in[5];
  const float* adj   = (const float*)d_in[6];
  const float* nnb   = (const float*)d_in[7];
  const float* Wg    = (const float*)d_in[8];
  const float* bg    = (const float*)d_in[9];
  const float* Wc    = (const float*)d_in[10];
  const float* bc    = (const float*)d_in[11];
  const float* Wp    = (const float*)d_in[12];
  const float* bp    = (const float*)d_in[13];
  const float* Wo    = (const float*)d_in[14];
  const float* bo    = (const float*)d_in[15];

  char* p = (char*)d_ws;
  auto alloc = [&p](size_t bytes) -> char* {
    char* r = p;
    p += (bytes + 255) & ~(size_t)255;
    return r;
  };
  bf16* adj_b  = (bf16*)alloc((size_t)cNL * pNL * 2);
  bf16* avg_b  = (bf16*)alloc((size_t)cNL * pE * 2);
  bf16* avgT_b = (bf16*)alloc((size_t)cE * pNL * 2);
  bf16* agg_b  = (bf16*)alloc((size_t)cNL * pE * 2);
  float* gemb  = (float*)alloc((size_t)cNL * cFEAT * 4);
  bf16* WgT    = (bf16*)alloc((size_t)cH * pE * 2);
  bf16* WcT    = (bf16*)alloc((size_t)cK * cF * pE * 2);
  bf16* WpT    = (bf16*)alloc((size_t)cE * pF * 2);
  bf16* WoT    = (bf16*)alloc((size_t)cFEAT * pF * 2);
  bf16* xe     = (bf16*)alloc((size_t)cB * cL * pE * 2);
  bf16* c_b    = (bf16*)alloc((size_t)cB * cLP * pF * 2);
  bf16* cT_b   = (bf16*)alloc((size_t)cB * cF * pLP * 2);
  bf16* pj_b   = (bf16*)alloc((size_t)cB * cLP * pE * 2);
  bf16* attn_b = (bf16*)alloc((size_t)cB * cNL * pF * 2);

  float* logits = (float*)d_out;
  float* loss   = logits + (size_t)cB * cNL;

  // --- prep (padded/zeroed operands) ---
  cvt_adj_kernel<<<4096, 256, 0, stream>>>(adj, adj_b);
  transpose_pad_bf16_kernel<<<256, 256, 0, stream>>>(Wg, WgT, cH, cE, cH, pE);
  transpose_pad_bf16_kernel<<<256, 256, 0, stream>>>(Wp, WpT, cE, cF, cE, pF);
  transpose_pad_bf16_kernel<<<256, 256, 0, stream>>>(Wo, WoT, cFEAT, cF, cFEAT, pF);
  wcperm_kernel<<<1024, 256, 0, stream>>>(Wc, WcT);
  xe_build_kernel<<<4096, 256, 0, stream>>>(x, mask, wemb, xe);
  label_avg_kernel<<<4096, 256, 0, stream>>>(lidx, lmask, wemb, gemb, avg_b,
                                             avgT_b);
  zpad_kernel<<<256, 256, 0, stream>>>(avg_b, cNL, pE, cE);
  zpad_kernel<<<256, 256, 0, stream>>>(avgT_b, cE, pNL, cNL);
  zpad_kernel<<<256, 256, 0, stream>>>(agg_b, cNL, pE, cE);
  zpad_kernel<<<256, 256, 0, stream>>>(pj_b, (ll)cB * cLP, pE, cE);
  zpad_kernel<<<256, 256, 0, stream>>>(attn_b, (ll)cB * cNL, pF, cF);

  // --- graph encoder ---
  // agg = (adj @ avg) / num_neighbors   [NL, E]
  gemm_nt_kernel<<<dim3((cNL + 63) / 64, (cE + 63) / 64), 128, 0, stream>>>(
      adj_b, (ll)pNL, avgT_b, (ll)pNL, cNL, cE, pNL, 0, nnb, nullptr, nullptr,
      0, agg_b, (ll)pE);
  // gh = relu(agg @ Wg + bg) -> graph_emb[:, E:E+H]
  gemm_nt_kernel<<<dim3((cNL + 63) / 64, (cH + 63) / 64), 128, 0, stream>>>(
      agg_b, (ll)pE, WgT, (ll)pE, cNL, cH, pE, 1, nullptr, bg, gemb + cE,
      (ll)cFEAT, nullptr, 0);

  // --- document encoder ---
  conv_gemm_kernel<<<dim3((cLP + 63) / 64, (cF + 63) / 64, cB), 128, 0,
                     stream>>>(xe, WcT, bc, c_b, cT_b);
  // pj = tanh(c @ Wp + bp)  [B*Lp, E]
  gemm_nt_kernel<<<dim3((cB * cLP + 63) / 64, (cE + 63) / 64), 128, 0,
                   stream>>>(c_b, (ll)pF, WpT, (ll)pF, cB * cLP, cE, pF, 2,
                             nullptr, bp, nullptr, 0, pj_b, (ll)pE);

  // --- label-wise attention (fused flash) ---
  attn_flash_kernel<<<dim3((cNL + 31) / 32, cB), 64, 0, stream>>>(avg_b, pj_b,
                                                                  cT_b, attn_b);

  // --- output projection + logits + loss ---
  proj_logits_kernel<<<dim3((cB * cNL + 63) / 64), 128, 0, stream>>>(
      attn_b, WoT, bo, gemb, logits);
  loss_kernel<<<1, 256, 0, stream>>>(logits, y, loss);
}
